// LinearFeatureHead_46265387712928
// MI455X (gfx1250) — matrix operand order repaired and checked by reading, not compiled, not run on hardware
//
#include <hip/hip_runtime.h>

// ---- CDNA5 vector types ----
typedef _Float16 v8h  __attribute__((ext_vector_type(8)));
typedef _Float16 v16h __attribute__((ext_vector_type(16)));
typedef float    v8f  __attribute__((ext_vector_type(8)));

#define BATCH 32768
#define HDIM  256
#define NDIM  1024
#define MT    128        // M tile per block
#define NTBL  128        // N tile per block
#define LDA   264        // f16 row stride in LDS (256 + 8 pad -> 132 dwords, 4-bank rotate)

// D = sigmoid(x @ W^T + b) ; x:[B,H] f32, W:[N,H] f32, b:[N] f32, out:[B,N] f32
__global__ __launch_bounds__(256)
void linear_heads_sigmoid_wmma(const float* __restrict__ x,
                               const float* __restrict__ W,
                               const float* __restrict__ bias,
                               float* __restrict__ out) {
  extern __shared__ _Float16 smem[];
  _Float16* As = smem;               // [MT][LDA]   x tile, f16
  _Float16* Bs = smem + MT * LDA;    // [NTBL][LDA] W tile, f16

  const int tid = threadIdx.x;
  const int m0  = blockIdx.x * MT;    // batch-row base
  const int n0  = blockIdx.y * NTBL;  // head-column base

  // ---------- stage x tile: f32 global -> f16 LDS (row-major, padded) ----------
  {
    const float4* srcX = (const float4*)(x + (size_t)m0 * HDIM);
    const float4* srcW = (const float4*)(W + (size_t)n0 * HDIM);
#pragma unroll
    for (int i = 0; i < 32; ++i) {
      int t   = i * 256 + tid;
      int row = t >> 6;          // 0..127
      int q   = t & 63;          // float4 index within row (k = 4*q)
      float4 v = srcX[row * 64 + q];
      auto lo = __builtin_amdgcn_cvt_pkrtz(v.x, v.y);   // v_cvt_pk_rtz_f16_f32
      auto hi = __builtin_amdgcn_cvt_pkrtz(v.z, v.w);
      uint2 u;
      u.x = __builtin_bit_cast(unsigned, lo);
      u.y = __builtin_bit_cast(unsigned, hi);
      *(uint2*)&As[row * LDA + q * 4] = u;              // ds_store_b64
    }
#pragma unroll
    for (int i = 0; i < 32; ++i) {
      int t   = i * 256 + tid;
      int row = t >> 6;
      int q   = t & 63;
      float4 v = srcW[row * 64 + q];
      auto lo = __builtin_amdgcn_cvt_pkrtz(v.x, v.y);
      auto hi = __builtin_amdgcn_cvt_pkrtz(v.z, v.w);
      uint2 u;
      u.x = __builtin_bit_cast(unsigned, lo);
      u.y = __builtin_bit_cast(unsigned, hi);
      *(uint2*)&Bs[row * LDA + q * 4] = u;
    }
  }
  __syncthreads();

  // ---------- per-wave 32(M) x 64(N) sub-tile, K swept 0..256 in steps of 32 ----------
  const int lane = tid & 31;
  const int wv   = tid >> 5;       // 0..7
  const int half = lane >> 4;      // lane half selects K-group (ISA A/B layouts)
  const int l16  = lane & 15;      // A: row M ; B: col N ; C: col N
  const int wm   = wv & 3;         // 4 M-groups of 32 rows
  const int wn   = wv >> 2;        // 2 N-groups of 64 cols

  v8f c[2][4];
#pragma unroll
  for (int mt = 0; mt < 2; ++mt)
#pragma unroll
    for (int nt = 0; nt < 4; ++nt)
      c[mt][nt] = (v8f){0.f, 0.f, 0.f, 0.f, 0.f, 0.f, 0.f, 0.f};

#pragma unroll
  for (int k0 = 0; k0 < HDIM; k0 += 32) {
    // A fragment 16x32 f16: lane(r,half) needs K = half*8..+7 (v0-3) and 16+half*8..+7 (v4-7)
    v16h afrag[2];
#pragma unroll
    for (int mt = 0; mt < 2; ++mt) {
      const _Float16* p = &As[(wm * 32 + mt * 16 + l16) * LDA + k0 + half * 8];
      v8h lo = *(const v8h*)p;           // ds_load_b128
      v8h hi = *(const v8h*)(p + 16);    // ds_load_b128
      afrag[mt] = __builtin_shufflevector(lo, hi, 0,1,2,3,4,5,6,7,8,9,10,11,12,13,14,15);
    }
    // B fragment 32x16 f16: lane(n,half) needs K = half*8..+7 (v0-3) and 16+half*8..+7 (v4-7)
    v16h bfrag[4];
#pragma unroll
    for (int nt = 0; nt < 4; ++nt) {
      const _Float16* p = &Bs[(wn * 64 + nt * 16 + l16) * LDA + k0 + half * 8];
      v8h lo = *(const v8h*)p;
      v8h hi = *(const v8h*)(p + 16);
      bfrag[nt] = __builtin_shufflevector(lo, hi, 0,1,2,3,4,5,6,7,8,9,10,11,12,13,14,15);
    }
#pragma unroll
    for (int mt = 0; mt < 2; ++mt)
#pragma unroll
      for (int nt = 0; nt < 4; ++nt)
        c[mt][nt] = __builtin_amdgcn_wmma_f32_16x16x32_f16(
            false, afrag[mt], false, bfrag[nt], (short)0, c[mt][nt], false, false);
  }

  // ---------- epilogue: + bias, sigmoid, store f32 ----------
  // C/D layout: VGPR p, lanes0-15 -> M=p,N=lane ; lanes16-31 -> M=8+p,N=lane-16
#pragma unroll
  for (int nt = 0; nt < 4; ++nt) {
    const int col = n0 + wn * 64 + nt * 16 + l16;
    const float bv = bias[col];
#pragma unroll
    for (int mt = 0; mt < 2; ++mt) {
      const int rbase = m0 + wm * 32 + mt * 16 + half * 8;
      v8f acc = c[mt][nt];
#pragma unroll
      for (int p = 0; p < 8; ++p) {
        float t = acc[p] + bv;
        out[(size_t)(rbase + p) * NDIM + col] = 1.0f / (1.0f + __expf(-t));
      }
    }
  }
}

extern "C" void kernel_launch(void* const* d_in, const int* in_sizes, int n_in,
                              void* d_out, int out_size, void* d_ws, size_t ws_size,
                              hipStream_t stream) {
  (void)in_sizes; (void)n_in; (void)d_ws; (void)ws_size; (void)out_size;
  const float* x  = (const float*)d_in[0];
  const float* W  = (const float*)d_in[1];
  const float* b  = (const float*)d_in[2];
  float* out      = (float*)d_out;

  const size_t shmem = (size_t)(MT + NTBL) * LDA * sizeof(_Float16); // 132 KB
  // allow >64KB dynamic LDS (gfx1250 WGP has 320KB)
  hipFuncSetAttribute((const void*)linear_heads_sigmoid_wmma,
                      hipFuncAttributeMaxDynamicSharedMemorySize, (int)shmem);

  dim3 grid(BATCH / MT, NDIM / NTBL, 1);   // 256 x 8 blocks
  dim3 block(256, 1, 1);                   // 8 waves (wave32)
  linear_heads_sigmoid_wmma<<<grid, block, shmem, stream>>>(x, W, b, out);
}
